// CombineGraph_65025804861636
// MI455X (gfx1250) — matrix operand order repaired and checked by reading, not compiled, not run on hardware
//
#include <hip/hip_runtime.h>

// ---------- types & helpers -------------------------------------------------

typedef __attribute__((ext_vector_type(16))) __bf16 v16bf;
typedef __attribute__((ext_vector_type(8)))  float  v8f;

union FragB { v16bf v; unsigned short u[16]; };
union FragC { v8f   v; float          f[8];  };

__device__ __forceinline__ unsigned short f2bf(float f) {
  unsigned int u = __float_as_uint(f);
  unsigned int r = u + 0x7FFFu + ((u >> 16) & 1u);   // round-to-nearest-even
  return (unsigned short)(r >> 16);
}
__device__ __forceinline__ float bf2f(unsigned short h) {
  return __uint_as_float(((unsigned int)h) << 16);
}

// A-matrix (16x32 bf16) element->K mapping (contiguous, matches B):
// half=lane>>4; K = half*16 + e
__device__ __forceinline__ int kofA(int e, int half) {
  return (half << 4) + e;
}
// B-matrix (32x16 bf16): lanes 0-15 hold K=0..15 (VGPR v -> K=2v,2v+1),
// lanes 16-31 hold K=16..31.  K = half*16 + e ; N = lane&15.

__device__ __forceinline__ v8f wmma_bf16(const FragB& a, const FragB& b, v8f c) {
  return __builtin_amdgcn_wmma_f32_16x16x32_bf16(
      /*neg_a=*/false, a.v, /*neg_b=*/false, b.v,
      /*c_mod=*/(short)0, c, /*reuse_a=*/false, /*reuse_b=*/false);
}

// ---------- CDNA5 async global->LDS DMA (ASYNCcnt tracked) ------------------
// GV mode: LDS address comes from the VDST VGPR; low 32 bits of a generic pointer
// to __shared__ are the LDS byte offset.

#define USE_ASYNC_COPY 1

#if USE_ASYNC_COPY
__device__ __forceinline__ void async_copy16(void* lds_dst, const void* gsrc) {
  unsigned ldsoff = (unsigned)(size_t)lds_dst;
  asm volatile("global_load_async_to_lds_b128 %0, %1, off"
               :: "v"(ldsoff), "v"(gsrc) : "memory");
}
__device__ __forceinline__ void async_wait0() {
  asm volatile("s_wait_asynccnt 0x0" ::: "memory");
}
#endif

// ---------- problem constants ----------------------------------------------

#define Bsz 256
#define Dsz 256
#define Nn  100
#define N2n 50
#define NTn 150
#define Mrows (Bsz * N2n)   // 12800

// ---------- kernel: gather hidden = emb[items_ID] --------------------------

__global__ __launch_bounds__(256) void gather_hidden_kernel(
    const float* __restrict__ emb, const int* __restrict__ itemsID,
    float* __restrict__ HidF, unsigned short* __restrict__ testH) {
  int idx = blockIdx.x * 256 + threadIdx.x;
  if (idx >= Mrows * Dsz) return;
  int row = idx >> 8, d = idx & 255;
  float v = emb[(size_t)itemsID[row] * Dsz + d];
  HidF[idx] = v;
  testH[(size_t)row * 768 + 512 + d] = f2bf(v);   // test[:,512:768] = hidden
}

// ---------- kernel: fold weights into G1/G2 (bf16) + fused biases ----------

__global__ __launch_bounds__(256) void prep_weights_kernel(
    const float* __restrict__ w_ih, const float* __restrict__ w_hh,
    const float* __restrict__ W_l1, const float* __restrict__ W_l2,
    const float* __restrict__ b_ih, const float* __restrict__ b_hh,
    const float* __restrict__ b_l1, const float* __restrict__ b_l2,
    const float* __restrict__ W_in, const float* __restrict__ W_out,
    unsigned short* __restrict__ G1H, unsigned short* __restrict__ G2H,
    float* __restrict__ bias1, float* __restrict__ bias2,
    unsigned short* __restrict__ WinH, unsigned short* __restrict__ WoutH) {
  const int NG = 768 * 768;
  int i = blockIdx.x * 256 + threadIdx.x;
  if (i < NG) {
    int n = i / 768, k = i % 768;
    float v = W_l1[i] + (k < 512 ? w_ih[(size_t)n * 512 + k] : 0.f);
    G1H[i] = f2bf(v);
  } else if (i < 2 * NG) {
    int j = i - NG; int n = j / 768, k = j % 768;
    float v = W_l2[j] + (k >= 512 ? w_hh[(size_t)n * 256 + (k - 512)] : 0.f);
    G2H[j] = f2bf(v);
  } else if (i < 2 * NG + 768) {
    int j = i - 2 * NG; bias1[j] = b_ih[j] + b_l1[j];
  } else if (i < 2 * NG + 1536) {
    int j = i - 2 * NG - 768; bias2[j] = b_hh[j] + b_l2[j];
  } else if (i < 2 * NG + 1536 + 65536) {
    int j = i - 2 * NG - 1536; WinH[j] = f2bf(W_in[j]);
  } else if (i < 2 * NG + 1536 + 131072) {
    int j = i - 2 * NG - 1536 - 65536; WoutH[j] = f2bf(W_out[j]);
  }
}

// ---------- kernel: generic WMMA GEMM  C = A(bf16)[M,K] * W(bf16)[N,K]^T + b
// M multiple of 128, N multiple of 64, K multiple of 32. 256 thr = 8 waves.
// Tiles staged via 16B async DMA to LDS (ASYNCcnt), then fed to WMMA.

__global__ __launch_bounds__(256) void gemm_bf16_kernel(
    const unsigned short* __restrict__ A, int lda,
    const unsigned short* __restrict__ W,
    const float* __restrict__ bias,
    float* __restrict__ outF, unsigned short* __restrict__ outH,
    int M, int K, int N) {
  __shared__ unsigned short Atile[128 * 32];
  __shared__ unsigned short Wtile[64 * 32];
  int tid = threadIdx.x, lane = tid & 31, wave = tid >> 5;
  int half = lane >> 4, m = lane & 15;
  int m0 = blockIdx.x * 128, n0 = blockIdx.y * 64;

  FragC acc[4];
  for (int t = 0; t < 4; ++t)
    for (int q = 0; q < 8; ++q) acc[t].f[q] = 0.f;

  for (int kb = 0; kb < K; kb += 32) {
    // stage A tile: 128 rows x 32 cols bf16, 16B chunks
    for (int i = tid; i < 128 * 4; i += 256) {
      int rr = i >> 2, c4 = i & 3;
      const uint4* src =
          reinterpret_cast<const uint4*>(A + (size_t)(m0 + rr) * lda + kb) + c4;
      uint4* dst = reinterpret_cast<uint4*>(&Atile[rr * 32]) + c4;
#if USE_ASYNC_COPY
      async_copy16(dst, src);
#else
      *dst = *src;
#endif
    }
    // stage W tile: 64 rows x 32 cols bf16
    for (int i = tid; i < 64 * 4; i += 256) {
      int rr = i >> 2, c4 = i & 3;
      const uint4* src =
          reinterpret_cast<const uint4*>(W + (size_t)(n0 + rr) * K + kb) + c4;
      uint4* dst = reinterpret_cast<uint4*>(&Wtile[rr * 32]) + c4;
#if USE_ASYNC_COPY
      async_copy16(dst, src);
#else
      *dst = *src;
#endif
    }
#if USE_ASYNC_COPY
    async_wait0();
#endif
    __syncthreads();
    FragB a;
#pragma unroll
    for (int e = 0; e < 16; ++e)
      a.u[e] = Atile[(wave * 16 + m) * 32 + kofA(e, half)];
#pragma unroll
    for (int t = 0; t < 4; ++t) {
      FragB bm;
#pragma unroll
      for (int e = 0; e < 16; ++e)
        bm.u[e] = Wtile[(t * 16 + m) * 32 + half * 16 + e];  // B[k][n] = W[n][k]
      acc[t].v = wmma_bf16(a, bm, acc[t].v);
    }
    __syncthreads();
  }
#pragma unroll
  for (int t = 0; t < 4; ++t) {
#pragma unroll
    for (int vg = 0; vg < 8; ++vg) {
      int gm = m0 + wave * 16 + vg + 8 * half;
      int gn = n0 + t * 16 + m;
      float v = acc[t].f[vg] + bias[gn];
      if (outF) outF[(size_t)gm * N + gn] = v;
      if (outH) outH[(size_t)gm * N + gn] = f2bf(v);
    }
  }
}

// ---------- kernel: per-batch adjacency matmuls ----------------------------
// input_in  = adjID[b][:, :50]  @ Xin[b]  + b_iah  -> test[:, 0:256]
// input_out = adjID[b][:, 50:]  @ Xout[b] + b_oah  -> test[:, 256:512]
// Dynamic LDS: Ain[64][64], Aout[64][64], Xin_s[64][256], Xout_s[64][256] bf16.

__global__ __launch_bounds__(128) void adjmm_kernel(
    const float* __restrict__ adjID,
    const unsigned short* __restrict__ XinH,
    const unsigned short* __restrict__ XoutH,
    const float* __restrict__ b_iah, const float* __restrict__ b_oah,
    unsigned short* __restrict__ testH) {
  extern __shared__ unsigned char smem_adj[];
  unsigned short* Ain    = (unsigned short*)smem_adj;   // 64*64
  unsigned short* Aout   = Ain + 64 * 64;               // 64*64
  unsigned short* Xin_s  = Aout + 64 * 64;              // 64*256
  unsigned short* Xout_s = Xin_s + 64 * 256;            // 64*256
  int b = blockIdx.x;
  int tid = threadIdx.x, lane = tid & 31, wave = tid >> 5;
  int half = lane >> 4, m = lane & 15;

  // stage adjacency halves as bf16 (zero-padded to 64x64), float2 granular
  for (int i = tid; i < 64 * 32; i += 128) {
    int j = i >> 5, k2 = i & 31;          // k = 2*k2
    float2 vi = make_float2(0.f, 0.f), vo = make_float2(0.f, 0.f);
    if (j < N2n && 2 * k2 < N2n) {
      const float* row = adjID + ((size_t)b * N2n + j) * (2 * N2n);
      vi = *reinterpret_cast<const float2*>(row + 2 * k2);
      vo = *reinterpret_cast<const float2*>(row + N2n + 2 * k2);
    }
    ushort2 si, so;
    si.x = f2bf(vi.x); si.y = f2bf(vi.y);
    so.x = f2bf(vo.x); so.y = f2bf(vo.y);
    *reinterpret_cast<ushort2*>(&Ain[j * 64 + 2 * k2])  = si;
    *reinterpret_cast<ushort2*>(&Aout[j * 64 + 2 * k2]) = so;
  }
  // stage X panels (rows >= 50 zero), 16B chunks
  for (int i = tid; i < 2 * 64 * 32; i += 128) {
    int which = i >> 11, rem = i & 2047;
    int k = rem >> 5, c4 = rem & 31;
    const unsigned short* X = which ? XoutH : XinH;
    unsigned short* Xs = which ? Xout_s : Xin_s;
    uint4 v = make_uint4(0u, 0u, 0u, 0u);
    if (k < N2n)
      v = *reinterpret_cast<const uint4*>(X + ((size_t)b * N2n + k) * Dsz + c4 * 8);
    *reinterpret_cast<uint4*>(&Xs[k * Dsz + c4 * 8]) = v;
  }
  __syncthreads();

  for (int phase = 0; phase < 2; ++phase) {
    const unsigned short* Xs = phase ? Xout_s : Xin_s;
    const unsigned short* Aq = phase ? Aout  : Ain;
    const float* bb = phase ? b_oah : b_iah;
    for (int nt = wave; nt < 16; nt += 4) {
      int ncol0 = nt * 16;
      for (int itile = 0; itile < 4; ++itile) {
        FragC acc; for (int q = 0; q < 8; ++q) acc.f[q] = 0.f;
        for (int kb = 0; kb < 64; kb += 32) {
          FragB a, bm;
#pragma unroll
          for (int e = 0; e < 16; ++e)
            a.u[e] = Aq[(itile * 16 + m) * 64 + kb + kofA(e, half)];
#pragma unroll
          for (int e = 0; e < 16; ++e) {
            int k = kb + half * 16 + e;
            bm.u[e] = Xs[k * Dsz + ncol0 + m];   // padded rows are zero
          }
          acc.v = wmma_bf16(a, bm, acc.v);
        }
#pragma unroll
        for (int vg = 0; vg < 8; ++vg) {
          int gj = itile * 16 + vg + 8 * half;
          if (gj < N2n) {
            int gn = ncol0 + m;
            float v = acc.f[vg] + bb[gn];
            testH[((size_t)b * N2n + gj) * 768 + phase * 256 + gn] = f2bf(v);
          }
        }
      }
    }
  }
}

// ---------- kernel: GRU gates -> h2 ----------------------------------------

__global__ __launch_bounds__(256) void gate_kernel(
    const float* __restrict__ gi, const float* __restrict__ gh,
    const float* __restrict__ HidF, float* __restrict__ outH2) {
  int idx = blockIdx.x * 256 + threadIdx.x;
  if (idx >= Mrows * Dsz) return;
  int row = idx >> 8, d = idx & 255;
  const float* gir = gi + (size_t)row * 768;
  const float* ghr = gh + (size_t)row * 768;
  float i_r = gir[d], i_i = gir[256 + d], i_n = gir[512 + d];
  float h_r = ghr[d], h_i = ghr[256 + d], h_n = ghr[512 + d];
  float rg = 1.f / (1.f + __expf(-(i_r + h_r)));
  float ig = 1.f / (1.f + __expf(-(i_i + h_i)));
  float ng = tanhf(i_n + rg * h_n);
  float hv = HidF[idx];
  outH2[idx] = ng + ig * (hv - ng);
}

// ---------- kernel: fused local_agg ----------------------------------------
// 128 threads = 4 waves, one wave per relation r for the Gram stage.
// Dynamic LDS: hAll[NPAD][256] bf16, hs[4][16][256] bf16,
//              alphaF[16][NPAD] f32, alphaH[16][NPAD] bf16.

template <int N, int NPAD>
__global__ __launch_bounds__(128) void local_agg_kernel(
    const float* __restrict__ emb, const int* __restrict__ ids,
    const int* __restrict__ adjm, const float* __restrict__ arel,
    float* __restrict__ out) {
  extern __shared__ unsigned char smem[];
  unsigned short* hAll = (unsigned short*)smem;             // NPAD*256
  unsigned short* hs   = hAll + NPAD * 256;                 // 4*16*256
  float*  alphaF = (float*)(hs + 4 * 16 * 256);             // 16*NPAD
  unsigned short* alphaH = (unsigned short*)(alphaF + 16 * NPAD);

  const int nIT = (N + 15) / 16;
  int b   = blockIdx.x / nIT;
  int it  = blockIdx.x % nIT;
  int tid = threadIdx.x;
  int lane = tid & 31, wave = tid >> 5;
  int half = lane >> 4, m = lane & 15;

  // 1) load h[b] panel as bf16 (rows >= N zero)
  for (int ve = tid; ve < NPAD * 64; ve += 128) {
    int j = ve >> 6, d4 = (ve & 63) << 2;
    float4 v = make_float4(0.f, 0.f, 0.f, 0.f);
    if (j < N) {
      int id = ids[b * N + j];
      v = *reinterpret_cast<const float4*>(emb + (size_t)id * Dsz + d4);
    }
    ushort4 pk;
    pk.x = f2bf(v.x); pk.y = f2bf(v.y); pk.z = f2bf(v.z); pk.w = f2bf(v.w);
    *reinterpret_cast<ushort4*>(&hAll[j * Dsz + d4]) = pk;
  }
  __syncthreads();

  // 2) hs[r][il][d] = h[it*16+il][d] * a[r][d]  +  init alphaF = NEG
  for (int i = tid; i < 4 * 16 * 64; i += 128) {
    int r = i >> 10, il = (i >> 6) & 15, d4 = (i & 63) << 2;
    ushort4 hv = *reinterpret_cast<const ushort4*>(&hAll[(it * 16 + il) * Dsz + d4]);
    float4  av = *reinterpret_cast<const float4*>(&arel[r * Dsz + d4]);
    ushort4 o;
    o.x = f2bf(bf2f(hv.x) * av.x);
    o.y = f2bf(bf2f(hv.y) * av.y);
    o.z = f2bf(bf2f(hv.z) * av.z);
    o.w = f2bf(bf2f(hv.w) * av.w);
    *reinterpret_cast<ushort4*>(&hs[(r * 16 + il) * Dsz + d4]) = o;
  }
  for (int i = tid; i < 4 * NPAD; i += 128)
    reinterpret_cast<float4*>(alphaF)[i] =
        make_float4(-9.0e15f, -9.0e15f, -9.0e15f, -9.0e15f);
  __syncthreads();

  // 3) e_r tile = hs_r @ h^T ; leaky-relu; select by adj into alphaF
  const int r = wave;
  for (int jt = 0; jt < NPAD / 16; ++jt) {
    FragC c; for (int q = 0; q < 8; ++q) c.f[q] = 0.f;
    for (int kb = 0; kb < Dsz; kb += 32) {
      FragB a, bm;
#pragma unroll
      for (int e = 0; e < 16; ++e)
        a.u[e] = hs[r * 4096 + m * 256 + kb + kofA(e, half)];
#pragma unroll
      for (int e = 0; e < 16; ++e)   // B[k][n] = h[jt*16+n][k]
        bm.u[e] = hAll[(jt * 16 + m) * Dsz + kb + half * 16 + e];
      c.v = wmma_bf16(a, bm, c.v);
    }
#pragma unroll
    for (int vg = 0; vg < 8; ++vg) {
      int iLoc = vg + 8 * half;
      int gi = it * 16 + iLoc;
      int gj = jt * 16 + m;
      if (gi < N && gj < N) {
        float v = c.f[vg];
        v = v > 0.f ? v : 0.2f * v;                 // leaky relu
        if (adjm[((size_t)b * N + gi) * N + gj] == r + 1)
          alphaF[iLoc * NPAD + gj] = v;             // disjoint across waves
      }
    }
  }
  __syncthreads();

  // 4) row softmax (8 lanes per row)
  {
    int row = tid >> 3, t8 = tid & 7;
    float* ar = alphaF + row * NPAD;
    float mx = -3.4e38f;
    for (int j = t8; j < N; j += 8) mx = fmaxf(mx, ar[j]);
    mx = fmaxf(mx, __shfl_xor(mx, 1, 32));
    mx = fmaxf(mx, __shfl_xor(mx, 2, 32));
    mx = fmaxf(mx, __shfl_xor(mx, 4, 32));
    float sm = 0.f;
    for (int j = t8; j < N; j += 8) {
      float e = __expf(ar[j] - mx); ar[j] = e; sm += e;
    }
    sm += __shfl_xor(sm, 1, 32);
    sm += __shfl_xor(sm, 2, 32);
    sm += __shfl_xor(sm, 4, 32);
    float inv = 1.f / sm;
    for (int j = t8; j < NPAD; j += 8)
      alphaH[row * NPAD + j] = f2bf(j < N ? ar[j] * inv : 0.f);
  }
  __syncthreads();

  // 5) out = alpha @ h  (wave handles 4 of 16 column tiles)
  FragC acc[4];
  for (int t = 0; t < 4; ++t)
    for (int q = 0; q < 8; ++q) acc[t].f[q] = 0.f;
  for (int kb = 0; kb < NPAD; kb += 32) {
    FragB a;
#pragma unroll
    for (int e = 0; e < 16; ++e)
      a.u[e] = alphaH[m * NPAD + kb + kofA(e, half)];
#pragma unroll
    for (int t = 0; t < 4; ++t) {
      int ncol0 = (wave * 4 + t) * 16;
      FragB bm;
#pragma unroll
      for (int e = 0; e < 16; ++e)  // B[k][n] = h[k][ncol0+n]
        bm.u[e] = hAll[(kb + half * 16 + e) * Dsz + ncol0 + m];
      acc[t].v = wmma_bf16(a, bm, acc[t].v);
    }
  }
#pragma unroll
  for (int t = 0; t < 4; ++t) {
    int ncol0 = (wave * 4 + t) * 16;
#pragma unroll
    for (int vg = 0; vg < 8; ++vg) {
      int gi = it * 16 + vg + 8 * half;
      if (gi < N)
        out[((size_t)b * N + gi) * Dsz + ncol0 + m] = acc[t].f[vg];
    }
  }
}

// ---------- launch ----------------------------------------------------------

extern "C" void kernel_launch(void* const* d_in, const int* in_sizes, int n_in,
                              void* d_out, int out_size, void* d_ws, size_t ws_size,
                              hipStream_t stream) {
  (void)in_sizes; (void)n_in; (void)out_size; (void)ws_size;
  const float* emb    = (const float*)d_in[0];
  const float* la1_a  = (const float*)d_in[1];
  const float* lam_a  = (const float*)d_in[2];
  const float* w_ih   = (const float*)d_in[3];
  const float* w_hh   = (const float*)d_in[4];
  const float* b_ih   = (const float*)d_in[5];
  const float* b_hh   = (const float*)d_in[6];
  const float* b_iah  = (const float*)d_in[7];
  const float* b_oah  = (const float*)d_in[8];
  const float* W_in   = (const float*)d_in[9];
  const float* b_in   = (const float*)d_in[10];
  const float* W_out  = (const float*)d_in[11];
  const float* b_out  = (const float*)d_in[12];
  const float* W_l1   = (const float*)d_in[13];
  const float* b_l1   = (const float*)d_in[14];
  const float* W_l2   = (const float*)d_in[15];
  const float* b_l2   = (const float*)d_in[16];
  const float* adjID  = (const float*)d_in[17];
  const int* inputs   = (const int*)d_in[18];
  const int* adjm     = (const int*)d_in[19];
  const int* itemsID  = (const int*)d_in[22];
  const int* totItems = (const int*)d_in[23];
  const int* totAdj   = (const int*)d_in[24];

  float* out = (float*)d_out;
  float* h1 = out;
  float* h2 = out + (size_t)Bsz * Nn * Dsz;
  float* hm = h2 + (size_t)Bsz * N2n * Dsz;

  char* ws = (char*)d_ws;
  size_t off = 0;
  auto alloc = [&](size_t bytes) -> char* {
    char* p = ws + off;
    off += (bytes + 255) & ~(size_t)255;
    return p;
  };
  float*          HidF  = (float*)alloc((size_t)Mrows * Dsz * 4);
  unsigned short* testH = (unsigned short*)alloc((size_t)Mrows * 768 * 2);
  unsigned short* G1H   = (unsigned short*)alloc((size_t)768 * 768 * 2);
  unsigned short* G2H   = (unsigned short*)alloc((size_t)768 * 768 * 2);
  float*          bias1 = (float*)alloc(768 * 4);
  float*          bias2 = (float*)alloc(768 * 4);
  unsigned short* WinH  = (unsigned short*)alloc((size_t)65536 * 2);
  unsigned short* WoutH = (unsigned short*)alloc((size_t)65536 * 2);
  unsigned short* XinH  = (unsigned short*)alloc((size_t)Mrows * Dsz * 2);
  unsigned short* XoutH = (unsigned short*)alloc((size_t)Mrows * Dsz * 2);
  float*          giF   = (float*)alloc((size_t)Mrows * 768 * 4);
  float*          ghF   = (float*)alloc((size_t)Mrows * 768 * 4);

  // hidden gather (+ test[:,512:768])
  gather_hidden_kernel<<<(Mrows * Dsz) / 256, 256, 0, stream>>>(
      emb, itemsID, HidF, testH);

  // weight folding
  int prepN = 2 * 768 * 768 + 1536 + 131072;
  prep_weights_kernel<<<(prepN + 255) / 256, 256, 0, stream>>>(
      w_ih, w_hh, W_l1, W_l2, b_ih, b_hh, b_l1, b_l2, W_in, W_out,
      G1H, G2H, bias1, bias2, WinH, WoutH);

  // X_in / X_out = hidden @ W^T + b  (bf16 outputs)
  dim3 g1(Mrows / 128, Dsz / 64);
  gemm_bf16_kernel<<<g1, 256, 0, stream>>>(testH + 512, 768, WinH,  b_in,
                                           nullptr, XinH,  Mrows, 256, 256);
  gemm_bf16_kernel<<<g1, 256, 0, stream>>>(testH + 512, 768, WoutH, b_out,
                                           nullptr, XoutH, Mrows, 256, 256);

  // adjacency matmuls -> test[:, 0:512]
  size_t smemAdj = (size_t)(64 * 64 * 2 + 64 * 256 * 2) * 2;  // 81920 B
  adjmm_kernel<<<Bsz, 128, smemAdj, stream>>>(adjID, XinH, XoutH,
                                              b_iah, b_oah, testH);

  // gi / gh = test @ G^T + bias
  dim3 g2(Mrows / 128, 768 / 64);
  gemm_bf16_kernel<<<g2, 256, 0, stream>>>(testH, 768, G1H, bias1,
                                           giF, nullptr, Mrows, 768, 768);
  gemm_bf16_kernel<<<g2, 256, 0, stream>>>(testH, 768, G2H, bias2,
                                           ghF, nullptr, Mrows, 768, 768);

  // gates -> h2
  gate_kernel<<<(Mrows * Dsz) / 256, 256, 0, stream>>>(giF, ghF, HidF, h2);

  // local aggregations -> h1, hm
  size_t smem100 = (size_t)128 * 256 * 2 + 4 * 16 * 256 * 2
                 + (size_t)16 * 128 * 4 + (size_t)16 * 128 * 2;
  local_agg_kernel<Nn, 128><<<Bsz * 7, 128, smem100, stream>>>(
      emb, inputs, adjm, la1_a, h1);
  size_t smem150 = (size_t)160 * 256 * 2 + 4 * 16 * 256 * 2
                 + (size_t)16 * 160 * 4 + (size_t)16 * 160 * 2;
  local_agg_kernel<NTn, 160><<<Bsz * 10, 128, smem150, stream>>>(
      emb, totItems, totAdj, lam_a, hm);
}
